// TokenReconstructionBlock_1752346657617
// MI455X (gfx1250) — matrix operand order repaired and checked by reading, not compiled, not run on hardware
//
#include <hip/hip_runtime.h>
#include <hip/hip_bf16.h>
#include <stdint.h>

#define B_   8
#define N_   4096
#define M_   1024
#define C_   768
#define KTOP 20
#define TEMP 0.01f
#define SROW 1536   // interleaved sfeat row: 768 hi bf16 + 768 lo bf16

typedef __bf16 bf16_t;
typedef __attribute__((ext_vector_type(16))) __bf16 v16bf;
typedef __attribute__((ext_vector_type(8)))  __bf16 v8bf;
typedef __attribute__((ext_vector_type(8)))  float  v8f;
typedef __attribute__((ext_vector_type(4)))  float  v4f;

// ---------------------------------------------------------------------------
// Prep 1: sfeat f32 -> split bf16, interleaved per row ([hi 768][lo 768]),
// plus row norms ||sfeat_m||^2. One wave per row, 8 rows per block.
// ---------------------------------------------------------------------------
__global__ __launch_bounds__(256) void prep_sfeat(const float* __restrict__ sfeat,
                                                  bf16_t* __restrict__ sfeatP,
                                                  float* __restrict__ normS) {
  int row  = blockIdx.x * 8 + (threadIdx.x >> 5);   // [0, B_*M_)
  int lane = threadIdx.x & 31;
  const float* src = sfeat + (size_t)row * C_;
  bf16_t* dst = sfeatP + (size_t)row * SROW;
  float s = 0.f;
  #pragma unroll 4
  for (int j = 0; j < C_ / 32; ++j) {
    float v = src[lane + 32 * j];
    s += v * v;
    bf16_t h = (bf16_t)v;
    dst[lane + 32 * j] = h;                          // hi half
    dst[C_ + lane + 32 * j] = (bf16_t)(v - (float)h); // lo half
  }
  for (int o = 16; o; o >>= 1) s += __shfl_xor(s, o, 32);
  if (lane == 0) normS[row] = s;
}

// ---------------------------------------------------------------------------
// Prep 2: x [b][m][c] f32 -> xT [b][c][m] bf16 (32x32 LDS tile transpose)
// grid = B_ * (M_/32) * (C_/32), block 256
// ---------------------------------------------------------------------------
__global__ __launch_bounds__(256) void prep_xT(const float* __restrict__ x,
                                               bf16_t* __restrict__ xT) {
  int bid = blockIdx.x;
  int b = bid / ((M_ / 32) * (C_ / 32));
  int r = bid % ((M_ / 32) * (C_ / 32));
  int m0 = (r / (C_ / 32)) * 32;
  int c0 = (r % (C_ / 32)) * 32;
  __shared__ bf16_t tile[32][33];
  #pragma unroll
  for (int j = 0; j < 4; ++j) {
    int idx = threadIdx.x + 256 * j;
    int mi = idx >> 5, ci = idx & 31;
    tile[mi][ci] = (bf16_t)x[((size_t)b * M_ + m0 + mi) * C_ + c0 + ci];
  }
  __syncthreads();
  #pragma unroll
  for (int j = 0; j < 4; ++j) {
    int idx = threadIdx.x + 256 * j;
    int ci = idx >> 5, mi = idx & 31;
    xT[((size_t)b * C_ + c0 + ci) * M_ + m0 + mi] = tile[mi][ci];
  }
}

// ---------------------------------------------------------------------------
// Fused: split-bf16 distance GEMM -> exp weight (LDS f32) -> top-K threshold +
// L2 normalize -> bf16 attention (LDS) -> GEMM2 -> out.
// One workgroup = batch b, 32 rows of N. 8 waves: 2 row-groups(16n) x 4 col
// groups. GEMM1: wave owns 16n x 256m (3 WMMAs/tile). GEMM2: 16n x 192c.
// ---------------------------------------------------------------------------
__global__ __launch_bounds__(256) void fused_tokenrec(
    const float* __restrict__ feat,
    const bf16_t* __restrict__ sfeatP,
    const bf16_t* __restrict__ xT,
    const float* __restrict__ normS,
    float* __restrict__ out) {
  __shared__ __align__(16) float  w_s[32 * 1024];     // 128 KB weight rows
  __shared__ __align__(16) bf16_t att_s[32 * 1024];   // 64 KB bf16 attention
  __shared__ __align__(16) float normS_s[1024];
  __shared__ float normF_s[32];
  __shared__ float pnorm[32][8];

  const int b   = blockIdx.x >> 7;           // / 128
  const int n0  = (blockIdx.x & 127) << 5;   // * 32
  const int tid = threadIdx.x;
  const int lane   = tid & 31;
  const int wid    = tid >> 5;
  const int laneHi = lane >> 4;      // 0/1
  const int lane15 = lane & 15;
  const int wrow = wid >> 2;         // 0..1  (row group of 16)
  const int wcol = wid & 3;          // 0..3
  const int s8  = laneHi * 8;        // A-matrix K-run start
  const int s16 = laneHi * 16;       // unused; B K-runs also start at s8

  // ---------------- Phase 0: norms into LDS ----------------
  // normS tile via async global->LDS DMA path (ASYNCcnt)
  {
    const float* gsrc = normS + b * M_;
    #pragma unroll
    for (int j = 0; j < 4; ++j) {
      unsigned ldsoff = (unsigned)(uintptr_t)(&normS_s[tid + 256 * j]);
      unsigned goff   = (unsigned)((tid + 256 * j) * 4);
      asm volatile("global_load_async_to_lds_b32 %0, %1, %2"
                   :
                   : "v"(ldsoff), "v"(goff), "s"(gsrc)
                   : "memory");
    }
  }
  {
    int rrow = tid >> 3;   // 0..31
    int part = tid & 7;    // 0..7 : 96 c-values each
    const float* fr = feat + ((size_t)b * N_ + n0 + rrow) * C_ + part * 96;
    float s = 0.f;
    #pragma unroll 4
    for (int j = 0; j < 96; j += 4) {
      v4f v = *(const v4f*)(fr + j);
      s += v[0] * v[0] + v[1] * v[1] + v[2] * v[2] + v[3] * v[3];
    }
    pnorm[rrow][part] = s;
  }
  asm volatile("s_wait_asynccnt 0" ::: "memory");
  __syncthreads();
  if (tid < 32) {
    float s = 0.f;
    #pragma unroll
    for (int j = 0; j < 8; ++j) s += pnorm[tid][j];
    normF_s[tid] = s;
  }
  __syncthreads();

  // ------------ Phase 1: GEMM1 (feat x sfeat^T), split-bf16 WMMA ------------
  const int mbase = wcol << 8;  // * 256
  v8f zero8 = {0.f, 0.f, 0.f, 0.f, 0.f, 0.f, 0.f, 0.f};
  v8f acc[16];
  #pragma unroll
  for (int t = 0; t < 16; ++t) acc[t] = zero8;

  const float* arow = feat + ((size_t)b * N_ + n0 + wrow * 16 + lane15) * C_;
  // single per-lane base; all tile/hi/lo offsets are immediates
  const bf16_t* bbase = sfeatP + (size_t)(b * M_ + mbase + lane15) * SROW + s8;

  for (int kc = 0; kc < C_; kc += 32) {
    // A 16x32 bf16: lane=row, elems 0..7 = K s8..s8+7, elems 8..15 = K s8+16..+23
    v16bf ah, al;
    {
      v4f r0 = *(const v4f*)(arow + kc + s8);
      v4f r1 = *(const v4f*)(arow + kc + s8 + 4);
      v4f r2 = *(const v4f*)(arow + kc + s8 + 16);
      v4f r3 = *(const v4f*)(arow + kc + s8 + 20);
      #pragma unroll
      for (int j = 0; j < 4; ++j) {
        float f0 = r0[j], f1 = r1[j], f2 = r2[j], f3 = r3[j];
        bf16_t h0 = (bf16_t)f0, h1 = (bf16_t)f1, h2 = (bf16_t)f2, h3 = (bf16_t)f3;
        ah[j] = h0;      al[j]      = (bf16_t)(f0 - (float)h0);
        ah[4 + j] = h1;  al[4 + j]  = (bf16_t)(f1 - (float)h1);
        ah[8 + j] = h2;  al[8 + j]  = (bf16_t)(f2 - (float)h2);
        ah[12 + j] = h3; al[12 + j] = (bf16_t)(f3 - (float)h3);
      }
    }
    // B 32x16 bf16 per 16-col subtile: lane=col(m), elems 0..7 = K s8..s8+7, elems 8..15 = K s8+16..+23
    const bf16_t* bk = bbase + kc;
    #pragma unroll
    for (int t = 0; t < 16; ++t) {
      const bf16_t* bp = bk + t * (16 * SROW);     // immediate tile offset
      v8bf h0 = *(const v8bf*)(bp);
      v8bf h1 = *(const v8bf*)(bp + 16);
      v8bf l0 = *(const v8bf*)(bp + C_);           // lo half, +1536 B
      v8bf l1 = *(const v8bf*)(bp + C_ + 16);
      v16bf bh = __builtin_shufflevector(h0, h1, 0, 1, 2, 3, 4, 5, 6, 7,
                                         8, 9, 10, 11, 12, 13, 14, 15);
      v16bf bl = __builtin_shufflevector(l0, l1, 0, 1, 2, 3, 4, 5, 6, 7,
                                         8, 9, 10, 11, 12, 13, 14, 15);
      acc[t] = __builtin_amdgcn_wmma_f32_16x16x32_bf16(
          false, al, false, bh, (short)0, acc[t], false, false);
      acc[t] = __builtin_amdgcn_wmma_f32_16x16x32_bf16(
          false, ah, false, bl, (short)0, acc[t], false, false);
      acc[t] = __builtin_amdgcn_wmma_f32_16x16x32_bf16(
          false, ah, false, bh, (short)0, acc[t], false, false);
    }
  }

  // finalize: ds = nF + nS - 2*dot, clamp, exp -> weight LDS
  #pragma unroll
  for (int t = 0; t < 16; ++t) {
    int m = mbase + t * 16 + lane15;
    float ns = normS_s[m];
    #pragma unroll
    for (int i = 0; i < 8; ++i) {
      int nl = wrow * 16 + i + laneHi * 8;      // C/D layout: row = i (+8 hi lanes)
      float d = fmaxf(normF_s[nl] + ns - 2.0f * acc[t][i], 0.0f);
      w_s[nl * 1024 + m] = __expf(-TEMP * d);
    }
  }
  __syncthreads();

  // ---------------- Phase 2: top-K threshold + L2 normalize ----------------
  // wave `wid` owns rows wid*4 .. wid*4+3; 32 values per lane in VGPRs.
  for (int q = 0; q < 4; ++q) {
    int r = wid * 4 + q;
    float v[32];
    #pragma unroll
    for (int i = 0; i < 32; ++i) v[i] = w_s[r * 1024 + lane + 32 * i];
    // weights are in (0,1]: positive-float bits are order-preserving.
    unsigned lo = 0u, hi = 0x3F800001u;  // (1.0f + 1ulp)
    for (int it = 0; it < 31; ++it) {
      unsigned mid = (lo + hi) >> 1;
      float fm = __uint_as_float(mid);
      int cnt = 0;
      #pragma unroll
      for (int i = 0; i < 32; ++i) cnt += (v[i] >= fm) ? 1 : 0;
      for (int o = 16; o; o >>= 1) cnt += __shfl_xor(cnt, o, 32);
      if (cnt >= KTOP) lo = mid; else hi = mid;   // invariant: count(>=lo) >= K
    }
    float thr = __uint_as_float(lo);              // == K-th largest (mink)
    float ss = 0.f;
    #pragma unroll
    for (int i = 0; i < 32; ++i) {
      float mval = (v[i] >= thr) ? v[i] : 0.f;
      v[i] = mval;
      ss += mval * mval;
    }
    for (int o = 16; o; o >>= 1) ss += __shfl_xor(ss, o, 32);
    float scale = 1.0f / fmaxf(sqrtf(ss), 1e-12f);
    #pragma unroll
    for (int i = 0; i < 32; ++i)
      att_s[r * 1024 + lane + 32 * i] = (bf16_t)(v[i] * scale);
  }
  __syncthreads();

  // ---------------- Phase 3: GEMM2 (attention x x), bf16 WMMA ----------------
  const int cbase = wcol * 192;   // 4 col groups x 192 = 768
  v8f acc2[12];
  #pragma unroll
  for (int t = 0; t < 12; ++t) acc2[t] = zero8;

  const int nlA = wrow * 16 + lane15;
  const bf16_t* xbase = xT + ((size_t)b * C_ + cbase + lane15) * M_ + s8;
  for (int mc = 0; mc < M_; mc += 32) {
    // A from LDS attention: runs at mc+s8 and mc+s8+16
    v8bf a0 = *(const v8bf*)(att_s + nlA * 1024 + mc + s8);
    v8bf a1 = *(const v8bf*)(att_s + nlA * 1024 + mc + s8 + 16);
    v16bf a2 = __builtin_shufflevector(a0, a1, 0, 1, 2, 3, 4, 5, 6, 7,
                                       8, 9, 10, 11, 12, 13, 14, 15);
    const bf16_t* xk = xbase + mc;
    #pragma unroll
    for (int t = 0; t < 12; ++t) {
      const bf16_t* bp = xk + t * (16 * M_);       // immediate tile offset
      v8bf b0 = *(const v8bf*)bp;
      v8bf b1 = *(const v8bf*)(bp + 16);
      v16bf bt = __builtin_shufflevector(b0, b1, 0, 1, 2, 3, 4, 5, 6, 7,
                                         8, 9, 10, 11, 12, 13, 14, 15);
      acc2[t] = __builtin_amdgcn_wmma_f32_16x16x32_bf16(
          false, a2, false, bt, (short)0, acc2[t], false, false);
    }
  }

  // store output f32 [B,N,C]
  #pragma unroll
  for (int t = 0; t < 12; ++t) {
    int c = cbase + t * 16 + lane15;
    #pragma unroll
    for (int i = 0; i < 8; ++i) {
      int nl = wrow * 16 + i + laneHi * 8;
      out[((size_t)b * N_ + n0 + nl) * C_ + c] = acc2[t][i];
    }
  }
}

// ---------------------------------------------------------------------------
extern "C" void kernel_launch(void* const* d_in, const int* in_sizes, int n_in,
                              void* d_out, int out_size, void* d_ws, size_t ws_size,
                              hipStream_t stream) {
  const float* x     = (const float*)d_in[0];  // [B,M,C]
  const float* feat  = (const float*)d_in[1];  // [B,N,C]
  const float* sfeat = (const float*)d_in[2];  // [B,M,C]
  float* out = (float*)d_out;                  // [B,N,C] f32

  // workspace layout
  bf16_t* sfeatP = (bf16_t*)d_ws;                            // B*M*1536 bf16
  bf16_t* xT     = sfeatP + (size_t)B_ * M_ * SROW;          // B*C*M bf16
  float*  normS  = (float*)(xT + (size_t)B_ * C_ * M_);      // B*M f32

  prep_sfeat<<<(B_ * M_) / 8, 256, 0, stream>>>(sfeat, sfeatP, normS);
  prep_xT<<<B_ * (M_ / 32) * (C_ / 32), 256, 0, stream>>>(x, xT);
  fused_tokenrec<<<B_ * (N_ / 32), 256, 0, stream>>>(feat, sfeatP, xT, normS, out);
}
